// DynamicFusionModule_87084756894435
// MI455X (gfx1250) — matrix operand order repaired and checked by reading, compiled, not run on hardware
//
#include <hip/hip_runtime.h>

// ---- problem constants (reference: DIM=192, HEADS=4, B=2, H=W=64) ----
#define DIMC   192
#define NTOK   4096
#define BATCH  2
#define NHEAD  4
#define HD     48
#define HDP    64            // head dim padded to multiple of 32 for WMMA K-loop
#define BN     (BATCH*NTOK)  // 8192 tokens total

typedef __bf16 bf16;
typedef bf16  v16bf __attribute__((ext_vector_type(16)));
typedef bf16  v8bf  __attribute__((ext_vector_type(8)));
typedef float v8f   __attribute__((ext_vector_type(8)));

// ---------------- WMMA fragment helpers (CDNA5 16x16x32 bf16) ----------------
// A (16x32, MxK): lane r (and r+16) hold row r. Lane-half hl selects K chunks:
//   vector elems 0..7  = K in [hl*16,    hl*16+8)
//   vector elems 8..15 = K in [hl*16+8,  hl*16+16)
static __device__ inline v16bf load_a_frag(const bf16* __restrict__ row, int k0, int hl) {
  union { v16bf v; v8bf h[2]; } u;
  u.h[0] = *(const v8bf*)(row + k0 + hl * 16);
  u.h[1] = *(const v8bf*)(row + k0 + hl * 16 + 8);
  return u.v;
}
// B (32x16, KxN): lane r holds column r; elems 0..15 = K in [hl*16, hl*16+16)
// (column data must be contiguous in memory: we store all B operands that way)
static __device__ inline v16bf load_b_frag(const bf16* __restrict__ col, int k0, int hl) {
  return *(const v16bf*)(col + k0 + hl * 16);
}
static __device__ inline v8f wmma_bf16(v16bf a, v16bf b, v8f c) {
  return __builtin_amdgcn_wmma_f32_16x16x32_bf16(false, a, false, b, (short)0, c, false, false);
}
// C/D (16x16 f32): elem i of v8f <-> (M = i + 8*hl, N = r)

// ---------------- small utility kernels ----------------
__global__ void zero_kernel(float* __restrict__ p, int n) {
  int i = blockIdx.x * 256 + threadIdx.x;
  if (i < n) p[i] = 0.f;
}
__global__ void convert_kernel(const float* __restrict__ s, bf16* __restrict__ d, int n) {
  int i = blockIdx.x * 256 + threadIdx.x;
  if (i < n) d[i] = (bf16)s[i];
}
__global__ void axpy_kernel(float* __restrict__ dst, const float* __restrict__ src, int n) {
  int i = blockIdx.x * 256 + threadIdx.x;
  if (i < n) dst[i] += src[i];
}

// mask[b,n] = gt[b,n] > mean_b(gt)
__global__ void mask_kernel(const float* __restrict__ gt, float* __restrict__ mask) {
  __shared__ float red[256];
  __shared__ float thr_s;
  int b = blockIdx.x, t = threadIdx.x;
  float s = 0.f;
  for (int i = t; i < NTOK; i += 256) s += gt[b * NTOK + i];
  red[t] = s;
  __syncthreads();
  for (int st = 128; st > 0; st >>= 1) { if (t < st) red[t] += red[t + st]; __syncthreads(); }
  if (t == 0) thr_s = red[0] * (1.f / NTOK);
  __syncthreads();
  float thr = thr_s;
  for (int i = t; i < NTOK; i += 256)
    mask[b * NTOK + i] = (gt[b * NTOK + i] > thr) ? 1.f : 0.f;
}

// LayerNorm over C=192 of x[b,n,c] where src is [B,C,N] (strided) plus optional
// row-major f32 addend [token,192]; bf16 output [token,192]. One wave per token.
__global__ void ln_kernel(const float* __restrict__ src, const float* __restrict__ add,
                          const float* __restrict__ g, const float* __restrict__ bt,
                          bf16* __restrict__ out) {
  int wave = threadIdx.x >> 5, lane = threadIdx.x & 31;
  int token = blockIdx.x * 8 + wave;
  int b = token >> 12, n = token & 4095;
  const float* sb = src + (size_t)b * DIMC * NTOK + n;
  float x[6];
  float s = 0.f;
#pragma unroll
  for (int j = 0; j < 6; j++) {
    int c = lane + 32 * j;
    float v = sb[(size_t)c * NTOK];
    if (add) v += add[(size_t)token * DIMC + c];
    x[j] = v; s += v;
  }
  s += __shfl_xor(s, 1, 32); s += __shfl_xor(s, 2, 32);
  s += __shfl_xor(s, 4, 32); s += __shfl_xor(s, 8, 32); s += __shfl_xor(s, 16, 32);
  float mean = s * (1.f / DIMC);
  float vs = 0.f;
#pragma unroll
  for (int j = 0; j < 6; j++) { float d = x[j] - mean; vs += d * d; }
  vs += __shfl_xor(vs, 1, 32); vs += __shfl_xor(vs, 2, 32);
  vs += __shfl_xor(vs, 4, 32); vs += __shfl_xor(vs, 8, 32); vs += __shfl_xor(vs, 16, 32);
  float rs = rsqrtf(vs * (1.f / DIMC) + 1e-5f);
#pragma unroll
  for (int j = 0; j < 6; j++) {
    int c = lane + 32 * j;
    out[(size_t)token * DIMC + c] = (bf16)((x[j] - mean) * rs * g[c] + bt[c]);
  }
}

// ---------------- generic WMMA GEMM: C[M,Nout] = A[M,K] @ W[Nout,K]^T + bias ----------------
// one wave computes one 16x16 output tile; block = 8 waves
template <bool OUT_F32, bool GELU, bool ACCUM>
__global__ void gemm_kernel(const bf16* __restrict__ A, const bf16* __restrict__ W,
                            const float* __restrict__ bias, void* __restrict__ Cp,
                            int Nout, int K) {
  int wave = threadIdx.x >> 5, lane = threadIdx.x & 31;
  int hl = lane >> 4, r = lane & 15;
  int tile = blockIdx.x * 8 + wave;
  int ntn = Nout >> 4;
  int tm = tile / ntn, tn = tile % ntn;
  const bf16* arow = A + (size_t)(tm * 16 + r) * K;
  const bf16* brow = W + (size_t)(tn * 16 + r) * K;
  v8f c = {0.f, 0.f, 0.f, 0.f, 0.f, 0.f, 0.f, 0.f};
  for (int k0 = 0; k0 < K; k0 += 32) {
    v16bf a = load_a_frag(arow, k0, hl);
    v16bf b = load_b_frag(brow, k0, hl);
    c = wmma_bf16(a, b, c);
  }
  float bv = bias[tn * 16 + r];
  float* cf = (float*)Cp;
  bf16* cb = (bf16*)Cp;
#pragma unroll
  for (int i = 0; i < 8; i++) {
    float v = c[i] + bv;
    if (GELU) v = 0.5f * v * (1.0f + erff(v * 0.7071067811865476f));
    size_t oidx = (size_t)(tm * 16 + i + 8 * hl) * Nout + tn * 16 + r;
    if (OUT_F32) { if (ACCUM) cf[oidx] += v; else cf[oidx] = v; }
    else cb[oidx] = (bf16)v;
  }
}

// ---------------- repack QKV [token,576] into attention-friendly layouts ----------------
// qh/kh: [B*NHEAD, NTOK, HDP] bf16 (d padded 48->64 with zeros)
__global__ void repack_qk_kernel(const bf16* __restrict__ qkv,
                                 bf16* __restrict__ qh, bf16* __restrict__ kh) {
  int idx = blockIdx.x * 256 + threadIdx.x;   // 8192 * 1024
  int t = idx >> 10;
  int inner = idx & 1023;
  int sel = inner >> 9;       // 0=q, 1=k
  int rest = inner & 511;
  int hh = rest >> 6;
  int dp = rest & 63;
  bf16 v = (bf16)0.f;
  if (dp < HD) v = qkv[(size_t)t * 576 + sel * DIMC + hh * HD + dp];
  int b = t >> 12, n = t & 4095;
  bf16* dst = sel ? kh : qh;
  dst[(((size_t)(b * NHEAD + hh)) * NTOK + n) * HDP + dp] = v;
}
// vT: [B*NHEAD, HD, NTOK] bf16 (transposed so attn@V B-fragments are contiguous)
__global__ void repack_v_kernel(const bf16* __restrict__ qkv, bf16* __restrict__ vT) {
  int idx = blockIdx.x * 256 + threadIdx.x;   // 8192 * 192
  int t = idx / DIMC;
  int o = idx - t * DIMC;
  int hh = o / HD, d = o - hh * HD;
  int b = t >> 12, n = t & 4095;
  vT[((size_t)(b * NHEAD + hh) * HD + d) * NTOK + n] = qkv[(size_t)t * 576 + 2 * DIMC + o];
}

// ---------------- flash attention: wave per 16-query tile, online softmax ----------------
__global__ void attn_kernel(const bf16* __restrict__ qh, const bf16* __restrict__ kh,
                            const bf16* __restrict__ vT, const float* __restrict__ mask,
                            bf16* __restrict__ ao) {
  __shared__ __align__(64) bf16 pshare[8][512];   // per-wave 16x32 P staging
  int wave = threadIdx.x >> 5, lane = threadIdx.x & 31;
  int hl = lane >> 4, r = lane & 15;
  int tile = blockIdx.x * 8 + wave;               // 2048 tiles total
  int qt = tile & 255;                            // NTOK/16 q-tiles
  int bh = tile >> 8;                             // b*NHEAD + h
  int b = bh >> 2, hh = bh & 3;
  const bf16* qrow = qh + ((size_t)bh * NTOK + qt * 16 + r) * HDP;
  v16bf qa0 = load_a_frag(qrow, 0, hl);
  v16bf qa1 = load_a_frag(qrow, 32, hl);
  float fM[8], fS[8];
  v8f acc0 = {0.f, 0.f, 0.f, 0.f, 0.f, 0.f, 0.f, 0.f};
  v8f acc1 = acc0, acc2 = acc0;
#pragma unroll
  for (int i = 0; i < 8; i++) { fM[i] = -3.0e38f; fS[i] = 0.f; }
  bf16* pb = pshare[wave];
  const float scale = 0.14433756729740643f;       // 48^-0.5
  const bf16* vbase = vT + (size_t)bh * HD * NTOK;
  for (int key0 = 0; key0 < NTOK; key0 += 32) {
    const bf16* kp0 = kh + ((size_t)bh * NTOK + key0 + r) * HDP;
    const bf16* kp1 = kp0 + 16 * HDP;
    v8f c0 = {0.f, 0.f, 0.f, 0.f, 0.f, 0.f, 0.f, 0.f};
    v8f c1 = c0;
    c0 = wmma_bf16(qa0, load_b_frag(kp0, 0, hl), c0);
    c0 = wmma_bf16(qa1, load_b_frag(kp0, 32, hl), c0);
    c1 = wmma_bf16(qa0, load_b_frag(kp1, 0, hl), c1);
    c1 = wmma_bf16(qa1, load_b_frag(kp1, 32, hl), c1);
    bool m0 = mask[b * NTOK + key0 + r] > 0.5f;
    bool m1 = mask[b * NTOK + key0 + 16 + r] > 0.5f;
    float l0[8], l1[8];
#pragma unroll
    for (int i = 0; i < 8; i++) {
      l0[i] = m0 ? c0[i] * scale : -1e9f;
      l1[i] = m1 ? c1[i] * scale : -1e9f;
    }
#pragma unroll
    for (int i = 0; i < 8; i++) {
      float v = fmaxf(l0[i], l1[i]);
      v = fmaxf(v, __shfl_xor(v, 1, 32)); v = fmaxf(v, __shfl_xor(v, 2, 32));
      v = fmaxf(v, __shfl_xor(v, 4, 32)); v = fmaxf(v, __shfl_xor(v, 8, 32));
      float nm = fmaxf(fM[i], v);
      float al = __expf(fM[i] - nm);
      fM[i] = nm;
      float p0 = __expf(l0[i] - nm);
      float p1 = __expf(l1[i] - nm);
      float rs = p0 + p1;
      rs += __shfl_xor(rs, 1, 32); rs += __shfl_xor(rs, 2, 32);
      rs += __shfl_xor(rs, 4, 32); rs += __shfl_xor(rs, 8, 32);
      fS[i] = fS[i] * al + rs;
      acc0[i] *= al; acc1[i] *= al; acc2[i] *= al;
      pb[(i + 8 * hl) * 32 + r] = (bf16)p0;        // C-layout -> row-major P in LDS
      pb[(i + 8 * hl) * 32 + 16 + r] = (bf16)p1;
    }
    asm volatile("" ::: "memory");                 // keep LDS store->load ordered
    v16bf pa = load_a_frag(pb + r * 32, 0, hl);    // reload as A-fragment (16x32 keys)
    asm volatile("" ::: "memory");
    acc0 = wmma_bf16(pa, load_b_frag(vbase + (size_t)(0 + r) * NTOK + key0, 0, hl), acc0);
    acc1 = wmma_bf16(pa, load_b_frag(vbase + (size_t)(16 + r) * NTOK + key0, 0, hl), acc1);
    acc2 = wmma_bf16(pa, load_b_frag(vbase + (size_t)(32 + r) * NTOK + key0, 0, hl), acc2);
  }
  size_t base = ((size_t)b * NTOK + qt * 16) * DIMC + hh * HD;
#pragma unroll
  for (int i = 0; i < 8; i++) {
    float inv = 1.0f / fS[i];
    size_t rowo = base + (size_t)(i + 8 * hl) * DIMC;
    ao[rowo + 0 + r]  = (bf16)(acc0[i] * inv);
    ao[rowo + 16 + r] = (bf16)(acc1[i] * inv);
    ao[rowo + 32 + r] = (bf16)(acc2[i] * inv);
  }
}

// ---------------- final: out[b,c,n] = f_ir + f_vis + mask * delta ----------------
__global__ void final_kernel(const float* __restrict__ f_ir, const float* __restrict__ f_vis,
                             const float* __restrict__ mask, const float* __restrict__ delta,
                             float* __restrict__ out) {
  int idx = blockIdx.x * 256 + threadIdx.x;       // B*C*N = 1572864
  if (idx == 0) out[BATCH * DIMC * NTOK] = 0.f;   // aux_loss
  int b = idx / (DIMC * NTOK);
  int rem = idx - b * (DIMC * NTOK);
  int c = rem / NTOK;
  int n = rem - c * NTOK;
  int token = b * NTOK + n;
  out[idx] = f_ir[idx] + f_vis[idx] + mask[token] * delta[(size_t)token * DIMC + c];
}

// ---------------- host orchestration ----------------
extern "C" void kernel_launch(void* const* d_in, const int* in_sizes, int n_in,
                              void* d_out, int out_size, void* d_ws, size_t ws_size,
                              hipStream_t stream) {
  const float* f_ir  = (const float*)d_in[0];
  const float* f_vis = (const float*)d_in[1];
  const float* gt    = (const float*)d_in[2];

  char* ws = (char*)d_ws;
  size_t off = 0;
  auto take = [&](size_t bytes) { size_t o = off; off += (bytes + 255) & ~(size_t)255; return o; };
  float* maskp = (float*)(ws + take(BATCH * NTOK * 4));
  bf16*  wb    = (bf16*) (ws + take((size_t)442368 * 2));     // in_w|out_w|w1|w2 bf16
  bf16*  qln   = (bf16*) (ws + take((size_t)BN * DIMC * 2));
  bf16*  kvln  = (bf16*) (ws + take((size_t)BN * DIMC * 2));
  bf16*  qkvp  = (bf16*) (ws + take((size_t)BN * 576 * 2));
  bf16*  qhp   = (bf16*) (ws + take((size_t)BATCH * NHEAD * NTOK * HDP * 2));
  bf16*  khp   = (bf16*) (ws + take((size_t)BATCH * NHEAD * NTOK * HDP * 2));
  bf16*  vTp   = (bf16*) (ws + take((size_t)BATCH * NHEAD * HD * NTOK * 2));
  bf16*  aop   = (bf16*) (ws + take((size_t)BN * DIMC * 2));
  float* attnp = (float*)(ws + take((size_t)BN * DIMC * 4));
  bf16*  hbuf  = (bf16*) (ws + take((size_t)BN * DIMC * 2));
  bf16*  gbuf  = (bf16*) (ws + take((size_t)BN * 768 * 2));
  float* delta = (float*)(ws + take((size_t)BN * DIMC * 4));
  (void)ws_size; (void)n_in; (void)in_sizes; (void)out_size;

  bf16* wb_in  = wb;
  bf16* wb_out = wb + 110592;
  bf16* wb_w1  = wb + 147456;
  bf16* wb_w2  = wb + 294912;

  mask_kernel<<<BATCH, 256, 0, stream>>>(gt, maskp);
  zero_kernel<<<(BN * DIMC) / 256, 256, 0, stream>>>(delta, BN * DIMC);

  for (int m = 0; m < 2; m++) {
    const float* P[14];
    for (int j = 0; j < 14; j++) P[j] = (const float*)d_in[3 + m * 14 + j];
    // 0 nq_g, 1 nq_b, 2 nkv_g, 3 nkv_b, 4 nffn_g, 5 nffn_b,
    // 6 in_w, 7 in_b, 8 out_w, 9 out_b, 10 w1, 11 b1, 12 w2, 13 b2
    const float* x_q  = (m == 0) ? f_ir : f_vis;
    const float* x_kv = (m == 0) ? f_vis : f_ir;

    convert_kernel<<<(110592 + 255) / 256, 256, 0, stream>>>(P[6], wb_in, 110592);
    convert_kernel<<<(36864 + 255) / 256, 256, 0, stream>>>(P[8], wb_out, 36864);
    convert_kernel<<<(147456 + 255) / 256, 256, 0, stream>>>(P[10], wb_w1, 147456);
    convert_kernel<<<(147456 + 255) / 256, 256, 0, stream>>>(P[12], wb_w2, 147456);

    ln_kernel<<<BN / 8, 256, 0, stream>>>(x_q, nullptr, P[0], P[1], qln);
    ln_kernel<<<BN / 8, 256, 0, stream>>>(x_kv, nullptr, P[2], P[3], kvln);

    // QKV: [8192,192] @ [576,192]^T -> bf16 [8192,576]
    gemm_kernel<false, false, false><<<(512 * 36) / 8, 256, 0, stream>>>(qln, wb_in, P[7], qkvp, 576, DIMC);
    repack_qk_kernel<<<(BN * 1024) / 256, 256, 0, stream>>>(qkvp, qhp, khp);
    repack_v_kernel<<<(BN * DIMC) / 256, 256, 0, stream>>>(qkvp, vTp);

    attn_kernel<<<(BATCH * NHEAD * (NTOK / 16)) / 8, 256, 0, stream>>>(qhp, khp, vTp, maskp, aop);

    // out proj -> f32 attn_out [8192,192]
    gemm_kernel<true, false, false><<<(512 * 12) / 8, 256, 0, stream>>>(aop, wb_out, P[9], attnp, DIMC, DIMC);
    axpy_kernel<<<(BN * DIMC) / 256, 256, 0, stream>>>(delta, attnp, BN * DIMC);

    // LN(x_q + attn_out) -> bf16
    ln_kernel<<<BN / 8, 256, 0, stream>>>(x_q, attnp, P[4], P[5], hbuf);
    // FFN1 + GELU -> bf16 [8192,768]
    gemm_kernel<false, true, false><<<(512 * 48) / 8, 256, 0, stream>>>(hbuf, wb_w1, P[11], gbuf, 768, DIMC);
    // FFN2 accumulates into delta (f32)
    gemm_kernel<true, false, true><<<(512 * 12) / 8, 256, 0, stream>>>(gbuf, wb_w2, P[13], delta, DIMC, 768);
  }

  final_kernel<<<(BATCH * DIMC * NTOK) / 256, 256, 0, stream>>>(f_ir, f_vis, maskp, delta, (float*)d_out);
}
